// SequentialDynamicMoE_56392920597063
// MI455X (gfx1250) — matrix operand order repaired and checked by reading, not compiled, not run on hardware
//
#include <hip/hip_runtime.h>
#include <hip/hip_bf16.h>

// Sequential dynamic MoE: B=131072 tokens, D=128, L=4 layers, O=1.
// Dense compute mapped onto v_wmma_f32_16x16x32_bf16 (f32 accumulation),
// async-to-LDS weight staging, 8 independent accumulator chains per wave.

typedef __bf16 v16bf __attribute__((ext_vector_type(16)));
typedef float  v8f   __attribute__((ext_vector_type(8)));

#define BT     128   // tokens per block
#define DM     128   // d_model
#define LAYERS 4
#define WHALF  32768 // 256*128 bf16 elements per layer

__device__ __forceinline__ unsigned short f2bf(float f) {
    unsigned int u = __float_as_uint(f);
    u += 0x7FFFu + ((u >> 16) & 1u);           // round-to-nearest-even
    return (unsigned short)(u >> 16);
}
__device__ __forceinline__ void unpack2(unsigned v, float& lo, float& hi) {
    lo = __uint_as_float(v << 16);
    hi = __uint_as_float(v & 0xffff0000u);
}

union FragU { v16bf v; uint4 q[2]; };

__device__ __forceinline__ v16bf load_frag2(const unsigned short* p0,
                                            const unsigned short* p1) {
    FragU u;
    u.q[0] = *(const uint4*)p0;   // ds_load_b128
    u.q[1] = *(const uint4*)p1;   // ds_load_b128
    return u.v;
}

// async 16B global -> LDS copy (GVS form: lds-addr VGPR, u32 offset, SGPR base)
__device__ __forceinline__ void async_copy16(unsigned ldsAddr, unsigned gOff,
                                             const void* base) {
    asm volatile("global_load_async_to_lds_b128 %0, %1, %2"
                 :: "v"(ldsAddr), "v"(gOff), "s"(base) : "memory");
}

// ---------------------------------------------------------------------------
// Prep: W_layers fp32 [4][256][128] -> bf16, pre-swizzled into the WMMA
// B-fragment lane layout: fragment (kstep, ntile); lane = 16*((k&31)>>4) +
// (n&15) holds 16 contiguous K-values (k&15) of its column n.
// ---------------------------------------------------------------------------
__global__ void prep_weights(const float* __restrict__ W,
                             unsigned short* __restrict__ wBf) {
    int e     = blockIdx.x * 256 + threadIdx.x;     // 0 .. 131071
    int depth = e >> 15;
    int rem   = e & (WHALF - 1);
    int k = rem >> 7, n = rem & 127;
    int kstep = k >> 5, kr = k & 31;
    int lane  = ((kr >> 4) & 1) * 16 + (n & 15);
    int h     = kr & 15;
    int idx   = ((kstep * 8 + (n >> 4)) * 32 + lane) * 16 + h;
    wBf[depth * WHALF + idx] = f2bf(W[e]);
}

// ---------------------------------------------------------------------------
// Main: one block = 128 tokens, 256 threads (8 waves). Each wave owns a
// 16-row strip; per layer: 8 k-steps x 8 n-tiles = 64 wmma, 8 acc chains.
// ---------------------------------------------------------------------------
__global__ __launch_bounds__(256)
void moe_main(const float* __restrict__ x, const float* __restrict__ Wr,
              const unsigned short* __restrict__ wBf,
              const float* __restrict__ bL, const float* __restrict__ wE,
              const float* __restrict__ bE, float* __restrict__ out) {
    extern __shared__ unsigned char smem[];
    unsigned short* sW   = (unsigned short*)smem;             // 64 KB swizzled W[depth]
    unsigned short* sX   = (unsigned short*)(smem + 65536);   // 32 KB x tile (bf16)
    unsigned short* sCur = (unsigned short*)(smem + 98304);   // 32 KB layer output (bf16)
    float* sWr   = (float*)(smem + 131072);                   // 2 KB router weights
    float* sBias = (float*)(smem + 133120);                   // 512 B layer bias
    float* sWest = (float*)(smem + 133632);                   // 512 B estimator col

    const int tid  = threadIdx.x;
    const int lane = tid & 31;
    const int wave = tid >> 5;
    const long tokenBase = (long)blockIdx.x * BT;
    const unsigned sWlds = (unsigned)(unsigned long long)(uintptr_t)sW;

    // ---- stage x tile fp32 -> bf16 (coalesced float4), stage Wr ----
    {
        const float4* xf4 = (const float4*)(x + tokenBase * DM);
        #pragma unroll
        for (int i = 0; i < 16; ++i) {
            int idx = i * 256 + tid;                 // 4096 float4 = 128x128 f32
            float4 v = xf4[idx];
            int row = idx >> 5;
            int k0  = (idx & 31) << 2;
            uint2 pk;
            pk.x = (unsigned)f2bf(v.x) | ((unsigned)f2bf(v.y) << 16);
            pk.y = (unsigned)f2bf(v.z) | ((unsigned)f2bf(v.w) << 16);
            *(uint2*)(sX + row * DM + k0) = pk;
        }
        #pragma unroll
        for (int i = 0; i < 2; ++i) {
            int idx = i * 256 + tid;                 // 512 floats of Wr [128][4]
            sWr[idx] = Wr[idx];
        }
    }
    __syncthreads();

    // ---- router gates: g = relu(x @ Wr), one token per thread (0..127) ----
    float g0 = 0.f, g1 = 0.f, g2 = 0.f, g3 = 0.f;
    if (tid < BT) {
        const uint4*  xr4 = (const uint4*)(sX + tid * DM);   // 16 x (8 bf16)
        const float4* wr4 = (const float4*)sWr;              // 128 x float4 rows
        #pragma unroll 4
        for (int i = 0; i < 16; ++i) {
            uint4 xq = xr4[i];
            unsigned wv[4] = {xq.x, xq.y, xq.z, xq.w};
            #pragma unroll
            for (int j = 0; j < 4; ++j) {
                float a, b; unpack2(wv[j], a, b);
                float4 wA = wr4[i * 8 + j * 2];
                float4 wB = wr4[i * 8 + j * 2 + 1];
                g0 += a * wA.x + b * wB.x;
                g1 += a * wA.y + b * wB.y;
                g2 += a * wA.z + b * wB.z;
                g3 += a * wA.w + b * wB.w;
            }
        }
        g0 = fmaxf(g0, 0.f); g1 = fmaxf(g1, 0.f);
        g2 = fmaxf(g2, 0.f); g3 = fmaxf(g3, 0.f);
    }

    float outAcc = 0.f, live = 1.f;

    #pragma unroll
    for (int depth = 0; depth < LAYERS; ++depth) {
        __syncthreads();   // previous layer's sW / sCur fully consumed

        // ---- async-stage swizzled W[depth] (64 KB) into LDS ----
        {
            const void* src = (const void*)(wBf + depth * WHALF);
            #pragma unroll
            for (int i = 0; i < 16; ++i) {
                unsigned idx = (unsigned)(i * 256 + tid) * 16u;   // byte offset
                async_copy16(sWlds + idx, idx, src);
            }
            if (tid < DM) {
                sBias[tid] = bL[depth * DM + tid];
                sWest[tid] = wE[depth * DM + tid];
            }
        }
        if (depth < LAYERS - 1)
            __builtin_prefetch(wBf + (depth + 1) * WHALF, 0, 1); // global_prefetch

        // ---- A fragments: this wave's 16 rows of concat[last, x] (K=256) ----
        // A k order used here (contiguous, matches B swizzle): lanes 0-15 hold K kb+0..15 (cb=0),
        // lanes 16-31 hold kb+16..31 (cb=16) -- contiguous 16B runs.
        const unsigned short* sLast = (depth == 0) ? sX : sCur;
        const int m  = wave * 16 + (lane & 15);
        const int cb = (lane >> 4) << 4;
        v16bf afrag[8];
        #pragma unroll
        for (int k = 0; k < 8; ++k) {
            const unsigned short* rowp = (k < 4)
                ? (sLast + m * DM + k * 32)          // K 0..127  = last_out
                : (sX    + m * DM + (k - 4) * 32);   // K 128..255 = x
            afrag[k] = load_frag2(rowp + cb, rowp + cb + 8);
        }
        asm volatile("s_wait_asynccnt 0x0" ::: "memory");
        __syncthreads();   // sW staged before B-fragment reads

        // ---- 8 independent accumulator chains (ntile inner) ----
        const int nIdx = lane & 15;
        const int hi8  = (lane >> 4) << 3;
        v8f acc[8];
        #pragma unroll
        for (int nt = 0; nt < 8; ++nt) acc[nt] = (v8f){};
        #pragma unroll
        for (int k = 0; k < 8; ++k) {
            #pragma unroll
            for (int nt = 0; nt < 8; ++nt) {
                const unsigned short* bp =
                    sW + ((unsigned)((k * 8 + nt) * 32 + lane)) * 16;
                v16bf bfrag = load_frag2(bp, bp + 8);
                acc[nt] = __builtin_amdgcn_wmma_f32_16x16x32_bf16(
                            false, afrag[k], false, bfrag,
                            (short)0, acc[nt], false, false);
            }
        }
        // ---- epilogue: bias + relu, f32 -> bf16 into sCur ----
        #pragma unroll
        for (int nt = 0; nt < 8; ++nt) {
            const int n = nt * 16 + nIdx;
            float bias = sBias[n];
            unsigned short* crow = sCur + (wave * 16 + hi8) * DM + n;
            #pragma unroll
            for (int j = 0; j < 8; ++j)
                crow[j * DM] = f2bf(fmaxf(acc[nt][j] + bias, 0.f));
        }
        __syncthreads();   // sCur complete for estimator pass

        // ---- estimator (128-dot) + ReLU-gate combine, one token/thread ----
        if (tid < BT) {
            float p = bE[depth];
            const uint4*  cr4 = (const uint4*)(sCur + tid * DM);
            const float4* we4 = (const float4*)sWest;
            #pragma unroll 4
            for (int i = 0; i < 16; ++i) {
                uint4 cq = cr4[i];
                float4 wa = we4[2 * i], wb = we4[2 * i + 1];
                float h0,h1,h2,h3,h4,h5,h6,h7;
                unpack2(cq.x, h0, h1); unpack2(cq.y, h2, h3);
                unpack2(cq.z, h4, h5); unpack2(cq.w, h6, h7);
                p += h0*wa.x + h1*wa.y + h2*wa.z + h3*wa.w
                   + h4*wb.x + h5*wb.y + h6*wb.z + h7*wb.w;
            }
            if (depth == LAYERS - 1) {
                outAcc += live * p;
            } else {
                float gd;  bool enter;
                if (depth == 0)      { gd = g0; enter = (g1 > 0.f) || (g2 > 0.f) || (g3 > 0.f); }
                else if (depth == 1) { gd = g1; enter = (g2 > 0.f) || (g3 > 0.f); }
                else                 { gd = g2; enter = (g3 > 0.f); }
                if (gd > 0.f) outAcc += live * p;   // mask_exit
                if (!enter)  live = 0.f;            // mask_enter prefix product
            }
        }
    }

    if (tid < BT) out[tokenBase + tid] = outAcc;
}

extern "C" void kernel_launch(void* const* d_in, const int* in_sizes, int n_in,
                              void* d_out, int out_size, void* d_ws, size_t ws_size,
                              hipStream_t stream) {
    const float* x  = (const float*)d_in[0];   // [131072,128]
    const float* Wr = (const float*)d_in[1];   // [128,4]
    const float* WL = (const float*)d_in[2];   // [4,256,128]
    const float* bL = (const float*)d_in[3];   // [4,128]
    const float* wE = (const float*)d_in[4];   // [4,128,1]
    const float* bE = (const float*)d_in[5];   // [4,1]
    float* out = (float*)d_out;                // [131072,1] fp32
    unsigned short* wBf = (unsigned short*)d_ws;   // 256 KB bf16 swizzled weights

    prep_weights<<<512, 256, 0, stream>>>(WL, wBf);

    const size_t ldsBytes = 134144;   // 131 KB: W 64K + x 32K + cur 32K + small
    moe_main<<<131072 / BT, 256, ldsBytes, stream>>>(x, Wr, wBf, bL, wE, bE, out);
}
